// DecoderModule_16853451669850
// MI455X (gfx1250) — compile-verified, matrix operand order checked by reading, not run on hardware
//
#include <hip/hip_runtime.h>
#include <hip/hip_bf16.h>

// ---------------------------------------------------------------------------
// Problem constants (from reference): H=16, D=128, E=2048, N=2048, F=8192
// ---------------------------------------------------------------------------
constexpr int kH  = 16;
constexpr int kD  = 128;
constexpr int kE  = 2048;
constexpr int kN  = 2048;
constexpr int kF  = 8192;
constexpr int kHD = kH * kD; // 2048

typedef __attribute__((ext_vector_type(16))) __bf16 v16bf;
typedef __attribute__((ext_vector_type(8)))  float  v8f;
typedef __attribute__((ext_vector_type(4)))  unsigned int v4u;
typedef __attribute__((ext_vector_type(8)))  int  v8i;
typedef __attribute__((ext_vector_type(4)))  int  v4i;

#if defined(__AMDGCN__) && __has_builtin(__builtin_amdgcn_tensor_load_to_lds) && \
    __has_builtin(__builtin_amdgcn_s_wait_tensorcnt)
#define HAS_TDM 1
#else
#define HAS_TDM 0
#endif

struct Pack32 { uint4 lo, hi; };

// Load a 16-element bf16 WMMA fragment from LDS.
// Caller passes pointer already offset to [row][koff]; per the CDNA5 ISA
// 16-bit A/B layout, elements 0..7 are k=koff..koff+7 and 8..15 are
// k=koff+16..koff+23 (lanes 16..31 use koff=+8, handled by caller).
__device__ __forceinline__ v16bf frag_ld(const __bf16* p) {
    Pack32 pk;
    pk.lo = *reinterpret_cast<const uint4*>(p);
    pk.hi = *reinterpret_cast<const uint4*>(p + 16);
    return __builtin_bit_cast(v16bf, pk);
}

__device__ __forceinline__ v8f wmma_bf16(v16bf a, v16bf b, v8f c) {
    return __builtin_amdgcn_wmma_f32_16x16x32_bf16(false, a, false, b,
                                                   (short)0, c, false, false);
}

#if HAS_TDM
// ---------------------------------------------------------------------------
// Tensor Data Mover: DMA a 2D tile [rows x cols_elems] of bf16 from global
// (row stride = row_stride_elems) into LDS at lds_byte_off, inserting 4
// DWORDs of padding after every 32 DWORDs stored (i.e. after each 64-elem
// row) -> LDS row stride of 72 bf16, matching the WMMA fragment layout.
// D# bitfields per CDNA5 ISA.
// ---------------------------------------------------------------------------
__device__ __forceinline__ void tdm_issue(v4u g0, v8i g1) {
    v4i z4 = {0, 0, 0, 0};
#if defined(__clang_major__) && (__clang_major__ >= 23)
    v8i z8 = {0, 0, 0, 0, 0, 0, 0, 0};
    __builtin_amdgcn_tensor_load_to_lds(g0, g1, z4, z4, z8, 0);
#else
    __builtin_amdgcn_tensor_load_to_lds(g0, g1, z4, z4, 0);
#endif
}

__device__ __forceinline__ void tdm_load_tile_2d(unsigned lds_byte_off,
                                                 const __bf16* gptr,
                                                 unsigned rows,
                                                 unsigned cols_elems,
                                                 unsigned row_stride_elems) {
    const unsigned long long ga = (unsigned long long)(uintptr_t)gptr;
    v4u g0;
    g0.x = 1u;                                   // count=1 (valid), user mode
    g0.y = lds_byte_off;                         // LDS dest (bytes)
    g0.z = (unsigned)ga;                         // global_addr[31:0]
    g0.w = (unsigned)((ga >> 32) & 0x01FFFFFFu)  // global_addr[56:32]
           | (2u << 30);                         // type=2 ("image")
    v8i g1;
    // workgroup_mask=0 | data_size=1 (2B) | pad_enable |
    // pad_interval=4 (32 DWORDs) | pad_amount=3 (4 DWORDs)
    g1[0] = (int)((1u << 16) | (1u << 20) | (4u << 22) | (3u << 25));
    const unsigned td0 = 0x7FFFFFFFu;            // huge tensor dims: no clip
    const unsigned td1 = 0x7FFFFFFFu;
    g1[1] = (int)((td0 & 0xFFFFu) << 16);        // abar_addr=0 | td0.lo16
    g1[2] = (int)(((td0 >> 16) & 0xFFFFu) | ((td1 & 0xFFFFu) << 16));
    g1[3] = (int)(((td1 >> 16) & 0xFFFFu) | ((cols_elems & 0xFFFFu) << 16));
    g1[4] = (int)(rows & 0xFFFFu);               // tile_dim1 | tile_dim2=0
    g1[5] = (int)row_stride_elems;               // tensor_dim0_stride lo32
    g1[6] = 0;                                   // stride hi16 | dim1_stride=0
    g1[7] = 0;
    tdm_issue(g0, g1);
}
#endif // HAS_TDM

// ---------------------------------------------------------------------------
// fp32 -> bf16 convert
// ---------------------------------------------------------------------------
__global__ __launch_bounds__(256)
void cvt_f32_to_bf16(const float* __restrict__ in, __bf16* __restrict__ out, long n) {
    long i = (long)blockIdx.x * blockDim.x + threadIdx.x;
    long stride = (long)gridDim.x * blockDim.x;
    for (; i < n; i += stride) out[i] = (__bf16)in[i];
}

// ---------------------------------------------------------------------------
// fp32 [R,C] -> bf16 [C,R] transposed convert (batched)
// ---------------------------------------------------------------------------
__global__ __launch_bounds__(256)
void transpose_cvt_bf16(const float* __restrict__ in, __bf16* __restrict__ out,
                        int R, int C, long inBatch, long outBatch) {
    __shared__ float t[32][33];
    const int z = blockIdx.z;
    in  += (long)z * inBatch;
    out += (long)z * outBatch;
    const int r0 = blockIdx.y << 5, c0 = blockIdx.x << 5;
    const int tx = threadIdx.x & 31, ty = threadIdx.x >> 5; // ty: 0..7
#pragma unroll
    for (int i = 0; i < 4; ++i) {
        int r = ty + i * 8;
        t[r][tx] = in[(long)(r0 + r) * C + c0 + tx];
    }
    __syncthreads();
#pragma unroll
    for (int i = 0; i < 4; ++i) {
        int c = ty + i * 8;
        out[(long)(c0 + c) * R + r0 + tx] = (__bf16)t[tx][c];
    }
}

// ---------------------------------------------------------------------------
// Tiled bf16 WMMA GEMM: C[M,Ndim] = A[M,K] * B^T   (B stored as [Ndim,K])
// Block tile 128x128, K-step 64. 256 threads = 8 waves, each wave 64x32.
// Tiles staged by the Tensor Data Mover (double-buffered, TENSORcnt-paced);
// falls back to cooperative vector loads when the TDM builtin is absent.
// ---------------------------------------------------------------------------
enum { EPI_F32 = 0, EPI_BF16 = 1, EPI_BIAS_RELU_BF16 = 2,
       EPI_BIAS_RES_F32 = 3, EPI_RES_F32 = 4 };

#define TILE_M 128
#define TILE_N 128
#define TILE_K 64
#define LDG   (TILE_K + 8)   // padded LDS stride (144B rows, 16B aligned)
#define TSZ   (TILE_M * LDG) // elements per tile buffer

constexpr size_t GEMM_SMEM = (size_t)4 * TSZ * 2;  // A0,A1,B0,B1 = 72KB

extern __shared__ char dyn_smem[];

template <int EPI>
__global__ __launch_bounds__(256)
void gemm_bf16_wmma(const __bf16* __restrict__ A, const __bf16* __restrict__ B,
                    float* __restrict__ Cf, __bf16* __restrict__ Cb,
                    const float* __restrict__ bias, const float* __restrict__ resid,
                    int M, int Ndim, int K,
                    long aBatch, long bBatch, long cBatch) {
    __bf16* As = (__bf16*)dyn_smem;              // [2][TSZ]
    __bf16* Bs = (__bf16*)dyn_smem + 2 * TSZ;    // [2][TSZ]

    const int tid  = threadIdx.x;
    const int lane = tid & 31;
    const int wid  = tid >> 5;
    const int wm   = wid >> 2;      // 0..1  (64 rows each)
    const int wn   = wid & 3;       // 0..3  (32 cols each)
    const int z    = blockIdx.z;
    A += (long)z * aBatch;
    B += (long)z * bBatch;
    const long cOff = (long)z * cBatch;
    const int m0 = blockIdx.y * TILE_M;
    const int n0 = blockIdx.x * TILE_N;

    const int rlo  = lane & 15;
    const int koff = (lane >> 4) << 3;   // 0 or 8, per 16-bit frag layout

    v8f acc[4][2];
#pragma unroll
    for (int i = 0; i < 4; ++i)
#pragma unroll
        for (int j = 0; j < 2; ++j) acc[i][j] = 0.f;

    auto compute_tiles = [&](const __bf16* Ab, const __bf16* Bb) {
#pragma unroll
        for (int ks = 0; ks < 2; ++ks) {
            v16bf af[4], bfg[2];
#pragma unroll
            for (int i = 0; i < 4; ++i)
                af[i] = frag_ld(&Ab[(wm * 64 + i * 16 + rlo) * LDG + ks * 32 + koff]);
#pragma unroll
            for (int j = 0; j < 2; ++j)
                bfg[j] = frag_ld(&Bb[(wn * 32 + j * 16 + rlo) * LDG + ks * 32 + koff]);
#pragma unroll
            for (int i = 0; i < 4; ++i)
#pragma unroll
                for (int j = 0; j < 2; ++j)
                    acc[i][j] = wmma_bf16(af[i], bfg[j], acc[i][j]);
        }
    };

#if HAS_TDM
    // -------- TDM double-buffered pipeline --------
    const __bf16* Abase = A + (long)m0 * K;
    const __bf16* Bbase = B + (long)n0 * K;
    int cur = 0;
    if (wid == 0) {
        tdm_load_tile_2d((unsigned)(uintptr_t)(As), Abase, TILE_M, TILE_K, K);
        tdm_load_tile_2d((unsigned)(uintptr_t)(Bs), Bbase, TILE_N, TILE_K, K);
    }
    for (int kb = 0; kb < K; kb += TILE_K) {
        if (wid == 0) {
            if (kb + TILE_K < K) {
                const int nxt = cur ^ 1;
                tdm_load_tile_2d((unsigned)(uintptr_t)(As + nxt * TSZ),
                                 Abase + kb + TILE_K, TILE_M, TILE_K, K);
                tdm_load_tile_2d((unsigned)(uintptr_t)(Bs + nxt * TSZ),
                                 Bbase + kb + TILE_K, TILE_N, TILE_K, K);
                // 2 newly issued outstanding; current buffer's 2 have landed
                __builtin_amdgcn_s_wait_tensorcnt(2);
            } else {
                __builtin_amdgcn_s_wait_tensorcnt(0);
            }
        }
        __syncthreads();
        compute_tiles(As + cur * TSZ, Bs + cur * TSZ);
        __syncthreads();   // all frag reads done before TDM overwrites buffer
        cur ^= 1;
    }
#else
    // -------- fallback: cooperative vector-load staging --------
    for (int kb = 0; kb < K; kb += TILE_K) {
#pragma unroll
        for (int u = 0; u < 4; ++u) {
            int li = tid + u * 256;       // 0..1023
            int r  = li >> 3;             // 8 uint4 per row
            int c  = (li & 7) * 8;        // element offset
            *reinterpret_cast<uint4*>(&As[r * LDG + c]) =
                *reinterpret_cast<const uint4*>(&A[(long)(m0 + r) * K + kb + c]);
            *reinterpret_cast<uint4*>(&Bs[r * LDG + c]) =
                *reinterpret_cast<const uint4*>(&B[(long)(n0 + r) * K + kb + c]);
        }
        if (kb + TILE_K < K) {
            __builtin_prefetch(&A[(long)(m0 + (tid >> 3)) * K + kb + TILE_K + (tid & 7) * 8], 0, 0);
            __builtin_prefetch(&B[(long)(n0 + (tid >> 3)) * K + kb + TILE_K + (tid & 7) * 8], 0, 0);
        }
        __syncthreads();
        compute_tiles(As, Bs);
        __syncthreads();
    }
#endif

    // Epilogue. C element layout: lane = {hi=lane>>4, col=lane&15}, row = v+8*hi.
    const int hi = lane >> 4, lo = lane & 15;
#pragma unroll
    for (int i = 0; i < 4; ++i) {
#pragma unroll
        for (int j = 0; j < 2; ++j) {
            const int colg = n0 + wn * 32 + j * 16 + lo;
#pragma unroll
            for (int v = 0; v < 8; ++v) {
                const int  rowg = m0 + wm * 64 + i * 16 + v + hi * 8;
                const long idx  = cOff + (long)rowg * Ndim + colg;
                float x = acc[i][j][v];
                if (EPI == EPI_BIAS_RELU_BF16) {
                    x += bias[colg];
                    x = x > 0.f ? x : 0.f;
                    Cb[idx] = (__bf16)x;
                } else if (EPI == EPI_BIAS_RES_F32) {
                    Cf[idx] = x + bias[colg] + resid[idx];
                } else if (EPI == EPI_RES_F32) {
                    Cf[idx] = x + resid[idx];
                } else if (EPI == EPI_BF16) {
                    Cb[idx] = (__bf16)x;
                } else {
                    Cf[idx] = x;
                }
            }
        }
    }
}

// ---------------------------------------------------------------------------
// Flash-style causal attention, one (head, 128-query block) per workgroup.
// Uses ~204KB of dynamic LDS (CDNA5 WGP has 320KB). All matmuls on WMMA.
// ---------------------------------------------------------------------------
#define AQ  128     // queries per block / keys per iteration
#define ADH 128     // head dim
#define LKB 136     // bf16 LDS stride (272B rows, 16B aligned)
#define LSF 132     // f32 LDS stride

constexpr size_t ATTN_SMEM =
    (size_t)4 * AQ * LKB * 2   // Qs, Ks, Vts, Ps  (bf16)
    + (size_t)AQ * LSF * 4     // Ss (f32 scores)
    + (size_t)3 * AQ * 4;      // mrow, lrow, crow

__global__ __launch_bounds__(256)
void attn_kernel(const __bf16* __restrict__ Q, const __bf16* __restrict__ K,
                 const __bf16* __restrict__ V, __bf16* __restrict__ Oc,
                 int Ntok, int Hh) {
    char* sp = dyn_smem;
    __bf16* Qs  = (__bf16*)sp; sp += AQ * LKB * 2;
    __bf16* Ks  = (__bf16*)sp; sp += AQ * LKB * 2;
    __bf16* Vts = (__bf16*)sp; sp += ADH * LKB * 2;   // V transposed: [d][key]
    __bf16* Ps  = (__bf16*)sp; sp += AQ * LKB * 2;
    float*  Ss  = (float*)sp;  sp += AQ * LSF * 4;
    float*  mrow = (float*)sp; sp += AQ * 4;
    float*  lrow = (float*)sp; sp += AQ * 4;
    float*  crow = (float*)sp;

    const int tid  = threadIdx.x;
    const int lane = tid & 31;
    const int wid  = tid >> 5;
    const int wm   = wid >> 2;    // 0..1
    const int wn   = wid & 3;     // 0..3
    const int rlo  = lane & 15;
    const int koff = (lane >> 4) << 3;
    const int hi   = lane >> 4, lo = lane & 15;

    const int h  = blockIdx.y;
    const int ib = blockIdx.x;
    const int q0 = ib * AQ;
    const __bf16* Qh = Q + ((long)h * Ntok + q0) * ADH;
    const __bf16* Kh = K + (long)h * Ntok * ADH;
    const __bf16* Vh = V + (long)h * Ntok * ADH;

    // Load 128x128 bf16 Q tile: 2048 uint4, 8 per thread.
#pragma unroll
    for (int u = 0; u < 8; ++u) {
        int li = tid + u * 256;
        int r  = li >> 4;
        int c  = (li & 15) * 8;
        *reinterpret_cast<uint4*>(&Qs[r * LKB + c]) =
            *reinterpret_cast<const uint4*>(&Qh[(long)r * ADH + c]);
    }
    if (tid < AQ) { mrow[tid] = -1e30f; lrow[tid] = 0.f; }

    v8f acc_o[4][2];
#pragma unroll
    for (int i = 0; i < 4; ++i)
#pragma unroll
        for (int j = 0; j < 2; ++j) acc_o[i][j] = 0.f;

    const float scale = 0.08838834764831845f;  // 1/sqrt(128)

    for (int jb = 0; jb <= ib; ++jb) {
        const __bf16* Kb = Kh + (long)jb * AQ * ADH;
        const __bf16* Vb = Vh + (long)jb * AQ * ADH;
        __syncthreads();  // previous iteration's WMMA frag reads are done
#pragma unroll
        for (int u = 0; u < 8; ++u) {
            int li = tid + u * 256;
            int r  = li >> 4;
            int c  = (li & 15) * 8;
            *reinterpret_cast<uint4*>(&Ks[r * LKB + c]) =
                *reinterpret_cast<const uint4*>(&Kb[(long)r * ADH + c]);
            uint4 vv = *reinterpret_cast<const uint4*>(&Vb[(long)r * ADH + c]);
            const __bf16* ve = reinterpret_cast<const __bf16*>(&vv);
#pragma unroll
            for (int e = 0; e < 8; ++e) Vts[(c + e) * LKB + r] = ve[e];
        }
        __syncthreads();

        // S = Q * K^T  (K tile rows are already the N x K layout WMMA wants)
        v8f as[4][2];
#pragma unroll
        for (int i = 0; i < 4; ++i)
#pragma unroll
            for (int j = 0; j < 2; ++j) as[i][j] = 0.f;
#pragma unroll
        for (int ks = 0; ks < 4; ++ks) {
            v16bf af[4], bfg[2];
#pragma unroll
            for (int i = 0; i < 4; ++i)
                af[i] = frag_ld(&Qs[(wm * 64 + i * 16 + rlo) * LKB + ks * 32 + koff]);
#pragma unroll
            for (int j = 0; j < 2; ++j)
                bfg[j] = frag_ld(&Ks[(wn * 32 + j * 16 + rlo) * LKB + ks * 32 + koff]);
#pragma unroll
            for (int i = 0; i < 4; ++i)
#pragma unroll
                for (int j = 0; j < 2; ++j)
                    as[i][j] = wmma_bf16(af[i], bfg[j], as[i][j]);
        }
        // Spill scaled scores to LDS for the row-parallel softmax.
#pragma unroll
        for (int i = 0; i < 4; ++i)
#pragma unroll
            for (int j = 0; j < 2; ++j)
#pragma unroll
                for (int v = 0; v < 8; ++v)
                    Ss[(wm * 64 + i * 16 + v + 8 * hi) * LSF + wn * 32 + j * 16 + lo] =
                        as[i][j][v] * scale;
        __syncthreads();

        // Online softmax: one thread per query row.
        if (tid < AQ) {
            const int r    = tid;
            const int qi   = q0 + r;
            const int kmax = min(AQ, qi - jb * AQ + 1);   // causal valid keys
            float* srow = &Ss[r * LSF];
            float  mold = mrow[r];
            float  mx   = mold;
            for (int c = 0; c < kmax; ++c) mx = fmaxf(mx, srow[c]);
            const float corr = __expf(mold - mx);
            float sum = 0.f;
            __bf16* prow = &Ps[r * LKB];
            for (int c = 0; c < AQ; ++c) {
                float e = (c < kmax) ? __expf(srow[c] - mx) : 0.f;
                prow[c] = (__bf16)e;
                sum += e;
            }
            lrow[r] = lrow[r] * corr + sum;
            mrow[r] = mx;
            crow[r] = corr;
        }
        __syncthreads();

        // Rescale running O, then O += P * V  (B = V^T rows = Vts).
#pragma unroll
        for (int i = 0; i < 4; ++i)
#pragma unroll
            for (int v = 0; v < 8; ++v) {
                const float f = crow[wm * 64 + i * 16 + v + 8 * hi];
#pragma unroll
                for (int j = 0; j < 2; ++j) acc_o[i][j][v] *= f;
            }
#pragma unroll
        for (int ks = 0; ks < 4; ++ks) {
            v16bf af[4], bfg[2];
#pragma unroll
            for (int i = 0; i < 4; ++i)
                af[i] = frag_ld(&Ps[(wm * 64 + i * 16 + rlo) * LKB + ks * 32 + koff]);
#pragma unroll
            for (int j = 0; j < 2; ++j)
                bfg[j] = frag_ld(&Vts[(wn * 32 + j * 16 + rlo) * LKB + ks * 32 + koff]);
#pragma unroll
            for (int i = 0; i < 4; ++i)
#pragma unroll
                for (int j = 0; j < 2; ++j)
                    acc_o[i][j] = wmma_bf16(af[i], bfg[j], acc_o[i][j]);
        }
    }

    // Normalize and write concat(heads) at [n, h*D + d] as bf16.
    const int HDall = Hh * ADH;
#pragma unroll
    for (int i = 0; i < 4; ++i)
#pragma unroll
        for (int j = 0; j < 2; ++j)
#pragma unroll
            for (int v = 0; v < 8; ++v) {
                const int r = wm * 64 + i * 16 + v + 8 * hi;
                const int d = wn * 32 + j * 16 + lo;
                const float inv = 1.f / lrow[r];
                Oc[(long)(q0 + r) * HDall + (long)h * ADH + d] =
                    (__bf16)(acc_o[i][j][v] * inv);
            }
}

// ---------------------------------------------------------------------------
// Row-wise LayerNorm over E columns, scalar gamma, vector beta.
// ---------------------------------------------------------------------------
__global__ __launch_bounds__(256)
void layernorm_kernel(const float* __restrict__ in, float* __restrict__ outF,
                      __bf16* __restrict__ outB, const float* __restrict__ gamma,
                      const float* __restrict__ beta, int Ecols) {
    __shared__ float rs[256], rs2[256];
    const int row = blockIdx.x;
    const int t   = threadIdx.x;
    const float* x = in + (long)row * Ecols;
    float s = 0.f, s2 = 0.f;
    for (int c = t; c < Ecols; c += 256) {
        float v = x[c];
        s += v;
        s2 += v * v;
    }
    rs[t] = s; rs2[t] = s2;
    __syncthreads();
    for (int o = 128; o > 0; o >>= 1) {
        if (t < o) { rs[t] += rs[t + o]; rs2[t] += rs2[t + o]; }
        __syncthreads();
    }
    const float mean = rs[0] / Ecols;
    const float var  = rs2[0] / Ecols - mean * mean;
    const float inv  = rsqrtf(var + 1e-5f) * gamma[0];
    for (int c = t; c < Ecols; c += 256) {
        float y = (x[c] - mean) * inv + beta[c];
        if (outF) outF[(long)row * Ecols + c] = y;
        if (outB) outB[(long)row * Ecols + c] = (__bf16)y;
    }
}

// ---------------------------------------------------------------------------
// Orchestration
// ---------------------------------------------------------------------------
extern "C" void kernel_launch(void* const* d_in, const int* in_sizes, int n_in,
                              void* d_out, int out_size, void* d_ws, size_t ws_size,
                              hipStream_t stream) {
    (void)in_sizes; (void)n_in; (void)out_size; (void)ws_size;

    const float* x     = (const float*)d_in[0];   // [N,E]
    const float* Wq    = (const float*)d_in[1];   // [H,E,D]
    const float* Wk    = (const float*)d_in[2];
    const float* Wv    = (const float*)d_in[3];
    const float* Wp    = (const float*)d_in[4];   // [HD,E]
    const float* W1    = (const float*)d_in[5];   // [E,F]
    const float* b1    = (const float*)d_in[6];   // [F]
    const float* W2    = (const float*)d_in[7];   // [F,E]
    const float* b2    = (const float*)d_in[8];   // [E]
    const float* gamma = (const float*)d_in[9];   // [1]
    const float* beta  = (const float*)d_in[10];  // [E]
    float* out = (float*)d_out;

    char* p = (char*)d_ws;
    auto alloc = [&](size_t bytes) -> char* {
        char* r = p;
        p += (bytes + 255) & ~(size_t)255;
        return r;
    };

    __bf16* xb     = (__bf16*)alloc((size_t)kN * kE * 2);
    __bf16* wqkvT  = (__bf16*)alloc((size_t)3 * kH * kD * kE * 2);   // [3][H][D][E]
    __bf16* qkvb   = (__bf16*)alloc((size_t)3 * kH * kN * kD * 2);   // [3][H][N][D]
    __bf16* poolT  = (__bf16*)alloc((size_t)kE * kHD * 2);           // [E][HD]
    __bf16* heads  = (__bf16*)alloc((size_t)kN * kHD * 2);           // [N][HD]
    float*  y1     = (float*) alloc((size_t)kN * kE * 4);
    float*  x1f    = (float*) alloc((size_t)kN * kE * 4);
    __bf16* x1b    = (__bf16*)alloc((size_t)kN * kE * 2);
    __bf16* w1T    = (__bf16*)alloc((size_t)kF * kE * 2);            // [F][E]
    __bf16* w2T    = (__bf16*)alloc((size_t)kE * kF * 2);            // [E][F]
    __bf16* h1b    = (__bf16*)alloc((size_t)kN * kF * 2);            // [N][F]
    float*  y2     = (float*) alloc((size_t)kN * kE * 4);

    const dim3 blk(256);

    // Allow >64KB dynamic LDS for the WMMA kernels (CDNA5 WGP: 320KB).
    (void)hipFuncSetAttribute((const void*)gemm_bf16_wmma<EPI_BF16>,
                              hipFuncAttributeMaxDynamicSharedMemorySize, (int)GEMM_SMEM);
    (void)hipFuncSetAttribute((const void*)gemm_bf16_wmma<EPI_RES_F32>,
                              hipFuncAttributeMaxDynamicSharedMemorySize, (int)GEMM_SMEM);
    (void)hipFuncSetAttribute((const void*)gemm_bf16_wmma<EPI_BIAS_RELU_BF16>,
                              hipFuncAttributeMaxDynamicSharedMemorySize, (int)GEMM_SMEM);
    (void)hipFuncSetAttribute((const void*)gemm_bf16_wmma<EPI_BIAS_RES_F32>,
                              hipFuncAttributeMaxDynamicSharedMemorySize, (int)GEMM_SMEM);
    (void)hipFuncSetAttribute((const void*)attn_kernel,
                              hipFuncAttributeMaxDynamicSharedMemorySize, (int)ATTN_SMEM);

    // --- stage inputs to bf16 (weights transposed to N-major for WMMA B) ---
    cvt_f32_to_bf16<<<4096, blk, 0, stream>>>(x, xb, (long)kN * kE);
    transpose_cvt_bf16<<<dim3(kD / 32, kE / 32, kH), blk, 0, stream>>>(
        Wq, wqkvT + (size_t)0 * kH * kD * kE, kE, kD, (long)kE * kD, (long)kD * kE);
    transpose_cvt_bf16<<<dim3(kD / 32, kE / 32, kH), blk, 0, stream>>>(
        Wk, wqkvT + (size_t)1 * kH * kD * kE, kE, kD, (long)kE * kD, (long)kD * kE);
    transpose_cvt_bf16<<<dim3(kD / 32, kE / 32, kH), blk, 0, stream>>>(
        Wv, wqkvT + (size_t)2 * kH * kD * kE, kE, kD, (long)kE * kD, (long)kD * kE);
    transpose_cvt_bf16<<<dim3(kE / 32, kHD / 32, 1), blk, 0, stream>>>(
        Wp, poolT, kHD, kE, 0, 0);
    transpose_cvt_bf16<<<dim3(kF / 32, kE / 32, 1), blk, 0, stream>>>(
        W1, w1T, kE, kF, 0, 0);
    transpose_cvt_bf16<<<dim3(kE / 32, kF / 32, 1), blk, 0, stream>>>(
        W2, w2T, kF, kE, 0, 0);

    // --- QKV projections: one batched GEMM over z = {q,k,v} x heads --------
    gemm_bf16_wmma<EPI_BF16><<<dim3(kD / 128, kN / 128, 3 * kH), blk, GEMM_SMEM, stream>>>(
        xb, wqkvT, nullptr, qkvb, nullptr, nullptr,
        kN, kD, kE, 0L, (long)kD * kE, (long)kN * kD);

    __bf16* qb = qkvb;
    __bf16* kb = qkvb + (size_t)kH * kN * kD;
    __bf16* vb = qkvb + (size_t)2 * kH * kN * kD;

    // --- causal flash attention (big dynamic LDS) --------------------------
    attn_kernel<<<dim3(kN / 128, kH), blk, ATTN_SMEM, stream>>>(
        qb, kb, vb, heads, kN, kH);

    // --- pool projection + residual ---------------------------------------
    gemm_bf16_wmma<EPI_RES_F32><<<dim3(kE / 128, kN / 128, 1), blk, GEMM_SMEM, stream>>>(
        heads, poolT, y1, nullptr, nullptr, x,
        kN, kE, kHD, 0L, 0L, 0L);

    // --- LayerNorm 1 -------------------------------------------------------
    layernorm_kernel<<<kN, blk, 0, stream>>>(y1, x1f, x1b, gamma, beta, kE);

    // --- FFN ---------------------------------------------------------------
    gemm_bf16_wmma<EPI_BIAS_RELU_BF16><<<dim3(kF / 128, kN / 128, 1), blk, GEMM_SMEM, stream>>>(
        x1b, w1T, nullptr, h1b, b1, nullptr,
        kN, kF, kE, 0L, 0L, 0L);
    gemm_bf16_wmma<EPI_BIAS_RES_F32><<<dim3(kE / 128, kN / 128, 1), blk, GEMM_SMEM, stream>>>(
        h1b, w2T, y2, nullptr, b2, x1f,
        kN, kE, kF, 0L, 0L, 0L);

    // --- LayerNorm 2 -> final output --------------------------------------
    layernorm_kernel<<<kN, blk, 0, stream>>>(y2, out, nullptr, gamma, beta, kE);
}
